// MultiConvLSTM_30717606101650
// MI455X (gfx1250) — matrix operand order repaired and checked by reading, not compiled, not run on hardware
//
#include <hip/hip_runtime.h>
#include <hip/hip_bf16.h>

// ---------------------------------------------------------------------------
// MultiConvLSTM on gfx1250: implicit-GEMM conv1d steps via v_wmma_f32_16x16x32_bf16
// with async global->LDS staging. B=8, T=64, Cin=64, L=256, F=128, K=5.
// ---------------------------------------------------------------------------

typedef __attribute__((ext_vector_type(16))) __bf16 v16bf;
typedef __attribute__((ext_vector_type(8)))  float  v8f;
typedef __attribute__((ext_vector_type(4)))  unsigned int u32x4;

__device__ __forceinline__ unsigned short f2bf(float f) {
    union { float f; unsigned int u; } v; v.f = f;
    unsigned int u = v.u;
    unsigned int r = u + 0x7FFFu + ((u >> 16) & 1u);   // round-to-nearest-even
    return (unsigned short)(r >> 16);
}
__device__ __forceinline__ float sigm(float x) {
    return 1.0f / (1.0f + __expf(-x));
}
__device__ __forceinline__ float tanh_fast(float x) {
    float e = __expf(-2.0f * fabsf(x));
    float t = (1.0f - e) / (1.0f + e);
    return x >= 0.0f ? t : -t;
}

// ------------------------- prep kernels ------------------------------------

// x [B,T,Cin,L] f32 -> xb [B,T,L,Cin] bf16 (l-major so LDS staging rows are contiguous)
__global__ void k_prep_x(const float* __restrict__ x, unsigned short* __restrict__ xb, int n) {
    int id = blockIdx.x * 256 + threadIdx.x;
    if (id >= n) return;
    int ci = id & 63;
    int l  = (id >> 6) & 255;
    int bt = id >> 14;
    xb[id] = f2bf(x[((size_t)bt * 64 + ci) * 256 + l]);
}

// h [B,F,L] f32 -> hb [B,L,F] bf16
__global__ void k_prep_h(const float* __restrict__ h, unsigned short* __restrict__ hb, int n) {
    int id = blockIdx.x * 256 + threadIdx.x;
    if (id >= n) return;
    int f = id & 127;
    int l = (id >> 7) & 255;
    int b = id >> 15;
    hb[id] = f2bf(h[((size_t)b * 128 + f) * 256 + l]);
}

__global__ void k_copy_f32(const float* __restrict__ src, float* __restrict__ dst, int n) {
    int id = blockIdx.x * 256 + threadIdx.x;
    if (id < n) dst[id] = src[id];
}

// Pack W [512, CTOT, 5] f32 into per-lane A-fragment-linear bf16 layout.
// K ordering: kk = k*CTOT + ci  (each 32-wide chunk lies inside one conv tap k).
// A 16x32 bf16 fragment layout (ISA 7.12.2): lane<16 holds K {0..7,16..23},
// lane>=16 holds K {8..15,24..31}; row M = lane&15.
template<int CTOT>
__global__ void k_prep_w(const float* __restrict__ W, unsigned short* __restrict__ Wr, int n) {
    constexpr int NCH = 5 * CTOT / 32;
    int id = blockIdx.x * 256 + threadIdx.x;
    if (id >= n) return;
    int j    = id & 15;
    int rest = id >> 4;
    int lane = rest & 31; rest >>= 5;
    int ch   = rest % NCH;
    int m16  = rest / NCH;
    int co   = m16 * 16 + (lane & 15);
    int half = lane >> 4;
    int kloc = (j < 8) ? (j + 8 * half) : (j + 8 + 8 * half);
    int kk   = ch * 32 + kloc;
    int k    = kk / CTOT;
    int ci   = kk % CTOT;
    Wr[id] = f2bf(W[((size_t)co * CTOT + ci) * 5 + k]);
}

// ------------------------- fused ConvLSTM step -----------------------------
// One time step of one layer. Grid: 4 M-groups x 32 N-blocks = 128 WGs, 256 thr.
// WG (mg, nb): batch b = nb>>2, columns l0..l0+63, channels f0 = mg*32 .. +31
// across all 4 gates (rows g*128 + f0 + 0..31). 8 waves: wave w -> gate w>>1,
// column-half w&1. Each wave: 2 A-tiles x 2 N-tiles = 4 accumulators, so each
// pair of A loads + pair of B loads feeds 4 WMMAs (LDS traffic halved per MAC).
template<int CTOT, int WA>
__global__ __launch_bounds__(256)
void k_step(const unsigned short* __restrict__ pA, long strA,   // part A of concat input, [.,l,ciA]
            const unsigned short* __restrict__ pB, long strB,   // h-state part, [b,l,128]
            const unsigned short* __restrict__ Wr,
            const float* __restrict__ bias,
            float* __restrict__ cst,                            // c state [b,f,l] f32 (in/out)
            unsigned short* __restrict__ hbOut,                 // h out [b,l,f] bf16
            float* __restrict__ outF)                           // optional f32 out [b,f,l]
{
    constexpr int PITCH = CTOT + 8;          // ushort elems; row stride 400/528 B (16B aligned, bank-spread)
    constexpr int ROWS  = 68;                // 64 cols + K=5 halo
    constexpr int NCH   = 5 * CTOT / 32;     // 30 (layer0) / 40 (layer1)
    constexpr int KB    = CTOT / 32;         // chunks per conv tap

    __shared__ __align__(16) unsigned short in_s[ROWS * PITCH];
    __shared__ float gbuf[4 * 32 * 65];      // [gate][f-row 0..31][col 0..63 (+pad)]

    const int tid = threadIdx.x;
    const int mg  = blockIdx.x >> 5;
    const int nb  = blockIdx.x & 31;
    const int b   = nb >> 2;
    const int l0  = (nb & 3) << 6;

    // ---- stage transposed bf16 input tile straight into LDS (async DMA) ----
    constexpr int VROW = CTOT / 8;           // 16B vectors per row
    for (int e = tid; e < ROWS * VROW; e += 256) {
        int lp = e / VROW;
        int ci = (e - lp * VROW) * 8;
        int l  = l0 - 2 + lp;
        unsigned short* dst = &in_s[lp * PITCH + ci];
        if (l >= 0 && l < 256) {
            const unsigned short* src = (ci < WA)
                ? (pA + (size_t)b * strA + (size_t)l * WA + ci)
                : (pB + (size_t)b * strB + (size_t)l * 128 + (ci - WA));
            unsigned lds_off = (unsigned)(size_t)dst;           // LDS aperture: low 32 bits
            unsigned long long gaddr = (unsigned long long)(size_t)src;
            asm volatile("global_load_async_to_lds_b128 %0, %1, off"
                         :: "v"(lds_off), "v"(gaddr) : "memory");
        } else {
            u32x4 z = {0u, 0u, 0u, 0u};
            *(u32x4*)dst = z;                                   // halo rows: zero pad
        }
    }
    asm volatile("s_wait_asynccnt 0x0" ::: "memory");
    __syncthreads();

    // ---- WMMA K-loop: 2x2 register tile per wave ----
    const int lane = tid & 31;
    const int w    = tid >> 5;
    const int g    = w >> 1;                 // gate 0..3
    const int nh   = w & 1;                  // column half 0..1
    const int m16a = g * 8 + mg * 2;         // first 16-row A tile
    const int half = lane >> 4;
    const int cl   = lane & 15;
    const int cb   = nh * 32;                // wave column base within WG tile

    v8f a00 = {0.f,0.f,0.f,0.f,0.f,0.f,0.f,0.f};  // A-tile0 x N-tile0
    v8f a01 = a00, a10 = a00, a11 = a00;

    const v16bf* wp0 = (const v16bf*)(Wr + ((size_t)m16a * NCH * 32 + lane) * 16);
    const v16bf* wp1 = (const v16bf*)(Wr + ((size_t)(m16a + 1) * NCH * 32 + lane) * 16);

    union Frag { u32x4 q[2]; v16bf v; };

#pragma unroll 5
    for (int ch = 0; ch < NCH; ++ch) {
        v16bf fa0 = wp0[ch * 32];                    // 32B per lane, L2-resident
        v16bf fa1 = wp1[ch * 32];
        int k   = ch / KB;
        int ci0 = (ch % KB) * 32 + half * 8;         // lane<16: K {0..7,16..23}, lane>=16: K {8..15,24..31}
        Frag fb0, fb1;
        const u32x4* r0 = (const u32x4*)&in_s[(cb + cl + k) * PITCH + ci0];
        fb0.q[0] = r0[0]; fb0.q[1] = r0[2];
        const u32x4* r1 = (const u32x4*)&in_s[(cb + 16 + cl + k) * PITCH + ci0];
        fb1.q[0] = r1[0]; fb1.q[1] = r1[2];
        a00 = __builtin_amdgcn_wmma_f32_16x16x32_bf16(false, fa0, false, fb0.v,
                                                      (short)0, a00, false, false);
        a01 = __builtin_amdgcn_wmma_f32_16x16x32_bf16(false, fa0, false, fb1.v,
                                                      (short)0, a01, false, false);
        a10 = __builtin_amdgcn_wmma_f32_16x16x32_bf16(false, fa1, false, fb0.v,
                                                      (short)0, a10, false, false);
        a11 = __builtin_amdgcn_wmma_f32_16x16x32_bf16(false, fa1, false, fb1.v,
                                                      (short)0, a11, false, false);
    }

    // ---- spill gate pre-activations (+bias) to LDS for cross-wave fusion ----
    const int cob = g * 128 + mg * 32;               // co of A-tile0 row 0
#pragma unroll
    for (int v = 0; v < 8; ++v) {
        int row = v + half * 8;                      // C/D layout: elem v, half -> M = v + 8*half
        float bv0 = bias[cob + row];
        float bv1 = bias[cob + 16 + row];
        gbuf[(g * 32 + row) * 65 + cb + cl]           = a00[v] + bv0;
        gbuf[(g * 32 + row) * 65 + cb + 16 + cl]      = a01[v] + bv0;
        gbuf[(g * 32 + 16 + row) * 65 + cb + cl]      = a10[v] + bv1;
        gbuf[(g * 32 + 16 + row) * 65 + cb + 16 + cl] = a11[v] + bv1;
    }
    __syncthreads();

    // ---- LSTM gate fusion: c,h update, 8 (f,l) cells per thread ----
    const int col = tid & 63;
    const int flb = tid >> 6;
#pragma unroll
    for (int i = 0; i < 8; ++i) {
        int fl = flb + i * 4;
        float ai = gbuf[(0 * 32 + fl) * 65 + col];
        float af = gbuf[(1 * 32 + fl) * 65 + col];
        float ao = gbuf[(2 * 32 + fl) * 65 + col];
        float ag = gbuf[(3 * 32 + fl) * 65 + col];
        int f = mg * 32 + fl;
        int l = l0 + col;
        size_t cidx = ((size_t)b * 128 + f) * 256 + l;
        float cOld = cst[cidx];
        float cNew = sigm(af) * cOld + sigm(ai) * tanh_fast(ag);
        float h    = sigm(ao) * tanh_fast(cNew);
        cst[cidx] = cNew;
        hbOut[((size_t)b * 256 + l) * 128 + f] = f2bf(h);
        if (outF) outF[cidx] = h;
    }
}

// ------------------------- host launcher -----------------------------------

extern "C" void kernel_launch(void* const* d_in, const int* in_sizes, int n_in,
                              void* d_out, int out_size, void* d_ws, size_t ws_size,
                              hipStream_t stream) {
    (void)in_sizes; (void)n_in; (void)out_size; (void)ws_size;
    const float* x  = (const float*)d_in[0];
    const float* W0 = (const float*)d_in[1];
    const float* b0 = (const float*)d_in[2];
    const float* W1 = (const float*)d_in[3];
    const float* b1 = (const float*)d_in[4];
    const float* h0 = (const float*)d_in[5];
    const float* c0 = (const float*)d_in[6];
    const float* h1 = (const float*)d_in[7];
    const float* c1 = (const float*)d_in[8];
    float* out = (float*)d_out;

    // workspace carve-up (256B aligned)
    char* ws = (char*)d_ws;
    size_t off = 0;
    auto take = [&](size_t bytes) { char* p = ws + off; off = (off + bytes + 255) & ~(size_t)255; return p; };
    unsigned short* xb   = (unsigned short*)take((size_t)8 * 64 * 256 * 64 * 2);   // 16 MB
    unsigned short* Wr0  = (unsigned short*)take((size_t)32 * 30 * 32 * 16 * 2);
    unsigned short* Wr1  = (unsigned short*)take((size_t)32 * 40 * 32 * 16 * 2);
    unsigned short* h0b[2], *h1b[2];
    h0b[0] = (unsigned short*)take((size_t)8 * 256 * 128 * 2);
    h0b[1] = (unsigned short*)take((size_t)8 * 256 * 128 * 2);
    h1b[0] = (unsigned short*)take((size_t)8 * 256 * 128 * 2);
    h1b[1] = (unsigned short*)take((size_t)8 * 256 * 128 * 2);
    float* c0s = (float*)take((size_t)8 * 128 * 256 * 4);
    float* c1s = (float*)take((size_t)8 * 128 * 256 * 4);

    // ---- one-time (per call) packing ----
    {
        int n = 8 * 64 * 256 * 64;
        k_prep_x<<<n / 256, 256, 0, stream>>>(x, xb, n);
    }
    {
        int n0 = 32 * 30 * 32 * 16;
        k_prep_w<192><<<n0 / 256, 256, 0, stream>>>(W0, Wr0, n0);
        int n1 = 32 * 40 * 32 * 16;
        k_prep_w<256><<<n1 / 256, 256, 0, stream>>>(W1, Wr1, n1);
    }
    {
        int n = 8 * 128 * 256;
        k_prep_h<<<n / 256, 256, 0, stream>>>(h0, h0b[0], n);
        k_prep_h<<<n / 256, 256, 0, stream>>>(h1, h1b[0], n);
        k_copy_f32<<<n / 256, 256, 0, stream>>>(c0, c0s, n);
        k_copy_f32<<<n / 256, 256, 0, stream>>>(c1, c1s, n);
    }

    // ---- 64 sequential ConvLSTM steps, 2 fused GEMM+gate kernels each ----
    const long strX = (long)64 * 256 * 64;    // batch stride of xb (elems)
    const long strH = (long)256 * 128;        // batch stride of h buffers (elems)
    for (int t = 0; t < 64; ++t) {
        int rd = t & 1, wr = (t + 1) & 1;
        // layer 0: input = concat(x[:,t], h0)   [CTOT=192, WA=64]
        k_step<192, 64><<<128, 256, 0, stream>>>(
            xb + (size_t)t * 256 * 64, strX,
            h0b[rd], strH,
            Wr0, b0, c0s, h0b[wr], (float*)nullptr);
        // layer 1: input = concat(hs0[t], h1)   [CTOT=256, WA=128]
        k_step<256, 128><<<128, 256, 0, stream>>>(
            h0b[wr], strH,
            h1b[rd], strH,
            Wr1, b1, c1s, h1b[wr],
            out + (size_t)t * 8 * 128 * 256);
    }
}
